// GptOssMlaAttention_63625645523667
// MI455X (gfx1250) — matrix operand order repaired and checked by reading, not compiled, not run on hardware
//
#include <hip/hip_runtime.h>
#include <hip/hip_bf16.h>
#include <math.h>

#define DEVFN __device__ __forceinline__

typedef __attribute__((ext_vector_type(16))) __bf16          v16bf;
typedef __attribute__((ext_vector_type(8)))  float           v8f;
typedef __attribute__((ext_vector_type(4)))  float           f4;
typedef __attribute__((ext_vector_type(8)))  unsigned short  u16x8;
typedef __attribute__((ext_vector_type(4)))  unsigned short  u16x4;

union FragU { v16bf v; u16x8 h[2]; };

constexpr int S_ = 2048, HID_ = 4096, H_ = 32;
constexpr int D_NOPE = 128, D_ROPE = 64, D_V = 128, RANK_ = 512;
constexpr int D_QK = 192, WINDOW_ = 128;
constexpr float SCALE_ = 0.07216878364870323f; // 192^-0.5

DEVFN unsigned short f2bf(float x) {
  unsigned u = __float_as_uint(x);
  return (unsigned short)((u + 0x7FFFu + ((u >> 16) & 1u)) >> 16);
}

// A-matrix fragment (16x32 bf16, MxK): lane row = lane&15; lanes<16 hold
// K {0..15}, lanes>=16 hold K {16..31} (matches frag_b K order).
DEVFN v16bf frag_a(const unsigned short* p, int ld) {
  int lane = threadIdx.x & 31;
  const unsigned short* r = p + (size_t)(lane & 15) * ld + (lane >> 4) * 16;
  FragU f;
  f.h[0] = *(const u16x8*)(r);
  f.h[1] = *(const u16x8*)(r + 8);
  return f.v;
}

// B-matrix fragment (32x16 bf16, KxN) read from K-contiguous storage T[n][k]:
// lane column = lane&15, K run = (lane>>4)*16 .. +15.
DEVFN v16bf frag_b(const unsigned short* p, int ld) {
  int lane = threadIdx.x & 31;
  const unsigned short* r = p + (size_t)(lane & 15) * ld + (lane >> 4) * 16;
  FragU f;
  f.h[0] = *(const u16x8*)(r);
  f.h[1] = *(const u16x8*)(r + 8);
  return f.v;
}

DEVFN v8f wmma_bf16(v16bf a, v16bf b, v8f c) {
  return __builtin_amdgcn_wmma_f32_16x16x32_bf16(false, a, false, b,
                                                 (short)0, c, false, false);
}

// Async global->LDS copy, 16 bytes per lane (GLOBAL_LOAD_ASYNC_TO_LDS_B128,
// GV mode, tracked by ASYNCcnt). lds_byte = per-lane LDS byte offset (low 32
// bits of the generic pointer per the LDS-aperture rule, ISA 10.2).
DEVFN void async_g2l_b128(unsigned lds_byte, unsigned long long gaddr) {
  asm volatile("global_load_async_to_lds_b128 %0, %1, off"
               :: "v"(lds_byte), "v"(gaddr)
               : "memory");
}
DEVFN void wait_async0() {
  asm volatile("s_wait_asynccnt 0" ::: "memory");
}

// ---------------------------------------------------------------------------
// One-time: fp32 [rows x cols] (row stride srcld) -> bf16 (row stride dstld).
// ---------------------------------------------------------------------------
__global__ __launch_bounds__(256)
void cvt_bf16_k(const float* __restrict__ src, unsigned short* __restrict__ dst,
                int cols4, int srcld, int dstld) {
  int idx = blockIdx.x * 256 + threadIdx.x;
  int r = idx / cols4, c4 = (idx % cols4) * 4;
  f4 v = *(const f4*)(src + (size_t)r * srcld + c4);
  u16x4 o;
#pragma unroll
  for (int j = 0; j < 4; j++) o[j] = f2bf(v[j]);
  *(u16x4*)(dst + (size_t)r * dstld + c4) = o;
}

// ---------------------------------------------------------------------------
// One-time: weight fp32 [K x N] -> bf16 transposed [N x K]. 64x64 LDS tiles.
// Requires K%64==0, N%64==0.
// ---------------------------------------------------------------------------
__global__ __launch_bounds__(256)
void transpose_cvt_k(const float* __restrict__ B, unsigned short* __restrict__ Bt,
                     int K, int N) {
  __shared__ unsigned short t[64 * 72];
  const int tid = threadIdx.x;
  const int n0 = blockIdx.x * 64, k0 = blockIdx.y * 64;
#pragma unroll
  for (int i = 0; i < 4; i++) {
    int id = tid + i * 256, kk = id >> 4, n4 = (id & 15) * 4;
    f4 v = *(const f4*)(B + (size_t)(k0 + kk) * N + n0 + n4);
#pragma unroll
    for (int j = 0; j < 4; j++) t[(n4 + j) * 72 + kk] = f2bf(v[j]);
  }
  __syncthreads();
#pragma unroll
  for (int i = 0; i < 2; i++) {
    int id = tid + i * 256, row = id >> 3, c8 = (id & 7) * 8;
    *(u16x8*)(Bt + (size_t)(n0 + row) * K + k0 + c8) = *(const u16x8*)(t + row * 72 + c8);
  }
}

// ---------------------------------------------------------------------------
// Tiled GEMM: C[M,N] = A[M,K] * B[K,N] (+bias). A bf16 [M][K]; B pre-
// transposed bf16 Bt [N][K]. BM=BN=128, BK=32; 8 waves (2x4); wave tile 64x32.
// LDS double buffer fed by async global->LDS copies (ASYNCcnt), one barrier
// per K-step. Requires M%128==0, K%32==0. For the ragged-N edge the B source
// row is clamped: B-tile row n only feeds output column n, and columns >= N
// are never stored, so the clamped (duplicate) data is harmless.
// ---------------------------------------------------------------------------
__global__ __launch_bounds__(256)
void gemm_k(const unsigned short* __restrict__ A, const unsigned short* __restrict__ Bt,
            const float* __restrict__ bias, float* __restrict__ C,
            int M, int N, int K, int lda, int ldb, int ldc) {
  __shared__ unsigned short sA[2 * 128 * 40];
  __shared__ unsigned short sB[2 * 128 * 40];
  const int tid = threadIdx.x;
  const int m0 = blockIdx.y * 128, n0 = blockIdx.x * 128;
  const int w = tid >> 5, wm = w >> 2, wn = w & 3;
  const int lane = tid & 31, lr = lane & 15, lh = lane >> 4;
  // staging coords: 512 chunks of 8 bf16 per tile -> 2 per thread per matrix
  const int row0 = tid >> 2, c8 = (tid & 3) * 8;
  const int row1 = row0 + 64;
  const int bn0 = min(n0 + row0, N - 1), bn1 = min(n0 + row1, N - 1);

  const unsigned la0 = (unsigned)(uintptr_t)&sA[row0 * 40 + c8];
  const unsigned la1 = (unsigned)(uintptr_t)&sA[row1 * 40 + c8];
  const unsigned lb0 = (unsigned)(uintptr_t)&sB[row0 * 40 + c8];
  const unsigned lb1 = (unsigned)(uintptr_t)&sB[row1 * 40 + c8];

  auto stage = [&](int k0, int buf) {
    const unsigned boff = (unsigned)buf * (128 * 40 * 2);
    async_g2l_b128(la0 + boff, (unsigned long long)(uintptr_t)(A + (size_t)(m0 + row0) * lda + k0 + c8));
    async_g2l_b128(la1 + boff, (unsigned long long)(uintptr_t)(A + (size_t)(m0 + row1) * lda + k0 + c8));
    async_g2l_b128(lb0 + boff, (unsigned long long)(uintptr_t)(Bt + (size_t)bn0 * ldb + k0 + c8));
    async_g2l_b128(lb1 + boff, (unsigned long long)(uintptr_t)(Bt + (size_t)bn1 * ldb + k0 + c8));
  };

  v8f acc[4][2];
#pragma unroll
  for (int i = 0; i < 4; i++)
#pragma unroll
    for (int j = 0; j < 2; j++) acc[i][j] = 0.0f;

  stage(0, 0);
  int it = 0;
  for (int k0 = 0; k0 < K; k0 += 32, ++it) {
    const int cur = it & 1;
    wait_async0();      // this wave's async copies into buffer `cur` landed
    __syncthreads();    // ...and everyone else's; also: all waves done reading `cur^1`
    if (k0 + 32 < K) {
      stage(k0 + 32, cur ^ 1);  // refill other buffer while we compute
      if (k0 + 64 < K)          // hint the tile after that (global_prefetch_b8)
        __builtin_prefetch(A + (size_t)(m0 + row0) * lda + k0 + 64, 0, 0);
    }
    const unsigned short* pA = sA + cur * (128 * 40);
    const unsigned short* pB = sB + cur * (128 * 40);
    v16bf af[4], bfr[2];
#pragma unroll
    for (int i = 0; i < 4; i++) af[i] = frag_a(pA + (wm * 64 + i * 16) * 40, 40);
#pragma unroll
    for (int j = 0; j < 2; j++) bfr[j] = frag_b(pB + (wn * 32 + j * 16) * 40, 40);
#pragma unroll
    for (int i = 0; i < 4; i++)
#pragma unroll
      for (int j = 0; j < 2; j++) acc[i][j] = wmma_bf16(af[i], bfr[j], acc[i][j]);
  }

  // epilogue: C layout = VGPR r -> row (r + half*8), col = lane&15
#pragma unroll
  for (int i = 0; i < 4; i++)
#pragma unroll
    for (int j = 0; j < 2; j++) {
      int cn = n0 + wn * 32 + j * 16 + lr;
      if (cn >= N) continue;
      float bv = bias ? bias[cn] : 0.0f;
#pragma unroll
      for (int r = 0; r < 8; r++) {
        int cm = m0 + wm * 64 + i * 16 + r + lh * 8;
        C[(size_t)cm * ldc + cn] = acc[i][j][r] + bv;
      }
    }
}

// ---------------------------------------------------------------------------
// RoPE + pack q (fp32 [S, H*192]) -> Q bf16 [H][S][192], SCALE folded in.
// ---------------------------------------------------------------------------
__global__ __launch_bounds__(256)
void rope_pack_q_k(const float* __restrict__ q, const float* __restrict__ cs,
                   const float* __restrict__ sn, unsigned short* __restrict__ Q) {
  int idx = blockIdx.x * 256 + threadIdx.x;
  int d = idx % D_QK;
  int t = idx / D_QK;
  int h = t % H_, s = t / H_;
  const float* row = q + (size_t)s * (H_ * D_QK) + h * D_QK;
  float val;
  if (d < D_NOPE) {
    val = row[d];
  } else {
    int dd = d - D_NOPE;
    float x = row[d];
    float rot = (dd < 32) ? -row[D_NOPE + dd + 32] : row[D_NOPE + dd - 32];
    val = x * cs[s * D_ROPE + dd] + rot * sn[s * D_ROPE + dd];
  }
  Q[((size_t)h * S_ + s) * D_QK + d] = f2bf(val * SCALE_);
}

// RoPE the shared k_rope part of ckv -> krope fp32 [S][64]
__global__ __launch_bounds__(256)
void rope_k_k(const float* __restrict__ ckv, const float* __restrict__ cs,
              const float* __restrict__ sn, float* __restrict__ krope) {
  int idx = blockIdx.x * 256 + threadIdx.x;
  int dd = idx % D_ROPE, s = idx / D_ROPE;
  const float* row = ckv + (size_t)s * (RANK_ + D_ROPE) + RANK_;
  float x = row[dd];
  float rot = (dd < 32) ? -row[dd + 32] : row[dd - 32];
  krope[idx] = x * cs[s * D_ROPE + dd] + rot * sn[s * D_ROPE + dd];
}

// Pack kv fp32 [S, H*256] + krope -> K bf16 [H][S][192], V bf16 [H][S][128]
__global__ __launch_bounds__(256)
void pack_kv_k(const float* __restrict__ kv, const float* __restrict__ krope,
               unsigned short* __restrict__ Kt, unsigned short* __restrict__ V) {
  int idx = blockIdx.x * 256 + threadIdx.x;
  int d = idx % 320;
  int t = idx / 320;
  int s = t % S_, h = t / S_;
  if (d < D_QK) {
    float v = (d < D_NOPE) ? kv[(size_t)s * (H_ * 256) + h * D_NOPE + d]
                           : krope[s * D_ROPE + (d - D_NOPE)];
    Kt[((size_t)h * S_ + s) * D_QK + d] = f2bf(v);
  } else {
    int dv = d - D_QK;
    V[((size_t)h * S_ + s) * D_V + dv] =
        f2bf(kv[(size_t)s * (H_ * 256) + H_ * D_NOPE + h * D_V + dv]);
  }
}

// ---------------------------------------------------------------------------
// Sliding-window attention with sinks. 1 wave per (16-query tile, head).
// Scores via 6x WMMA (K=192); online softmax; P*V via 8x WMMA (P zero-padded
// to 16x32, V^T staged in LDS). Output bf16 [S][H*128].
// ---------------------------------------------------------------------------
__global__ __launch_bounds__(32)
void attn_k(const unsigned short* __restrict__ Q, const unsigned short* __restrict__ Kt,
            const unsigned short* __restrict__ V, const float* __restrict__ sinks,
            unsigned short* __restrict__ O) {
  __shared__ unsigned short sP[16 * 40];    // P tile, K-padded to 32
  __shared__ unsigned short sVt[128 * 40];  // V^T: [dv][key]
  const int qt = blockIdx.x, h = blockIdx.y;
  const int q0 = qt * 16;
  const int lane = threadIdx.x & 31, lr = lane & 15, lh = lane >> 4;
  const unsigned short* Qh = Q + (size_t)h * S_ * D_QK;
  const unsigned short* Kh = Kt + (size_t)h * S_ * D_QK;
  const unsigned short* Vh = V + (size_t)h * S_ * D_V;

  v16bf qf[6];
#pragma unroll
  for (int c = 0; c < 6; ++c) qf[c] = frag_a(Qh + (size_t)q0 * D_QK + c * 32, D_QK);

  v8f acc[8];
#pragma unroll
  for (int j = 0; j < 8; ++j) acc[j] = 0.0f;
  float rmax[8], rsum[8];
#pragma unroll
  for (int r = 0; r < 8; ++r) { rmax[r] = -1e30f; rsum[r] = 0.0f; }

  // zero the padded K half of P once (never written again)
#pragma unroll
  for (int r = 0; r < 8; ++r) sP[(r + lh * 8) * 40 + 16 + lr] = 0;

  const int ktlo = (q0 >= WINDOW_) ? (q0 - (WINDOW_ - 1)) / 16 : 0;
  for (int kt = ktlo; kt <= qt; ++kt) {
    const int k0 = kt * 16;
    // ---- scores = Q (16x192) * K^T tile (192x16), bf16 WMMA
    v8f sc = 0.0f;
    const unsigned short* kr = Kh + (size_t)(k0 + lr) * D_QK + lh * 16;
#pragma unroll
    for (int c = 0; c < 6; ++c) {
      FragU f;
      f.h[0] = *(const u16x8*)(kr + c * 32);
      f.h[1] = *(const u16x8*)(kr + c * 32 + 8);
      sc = wmma_bf16(qf[c], f.v, sc);
    }
    // ---- mask + online softmax (16-lane reductions per half-wave)
    float corr[8];
#pragma unroll
    for (int r = 0; r < 8; ++r) {
      const int qi = q0 + r + lh * 8;
      const int kj = k0 + lr;
      float v = sc[r];
      if (!(kj <= qi && kj > qi - WINDOW_)) v = -__builtin_inff();
      float m = v;
#pragma unroll
      for (int off = 1; off < 16; off <<= 1) m = fmaxf(m, __shfl_xor(m, off, 16));
      float nm = fmaxf(rmax[r], m);
      float cr = __expf(rmax[r] - nm);
      float p = __expf(v - nm);
      float ps = p;
#pragma unroll
      for (int off = 1; off < 16; off <<= 1) ps += __shfl_xor(ps, off, 16);
      rsum[r] = rsum[r] * cr + ps;
      rmax[r] = nm;
      corr[r] = cr;
      sP[(r + lh * 8) * 40 + lr] = f2bf(p);
    }
#pragma unroll
    for (int j = 0; j < 8; ++j)
#pragma unroll
      for (int r = 0; r < 8; ++r) acc[j][r] *= corr[r];
    // ---- stage V^T (16 real key rows; padded K half of P is zero)
    {
      const unsigned short* vr = Vh + (size_t)(k0 + lr) * D_V + lh * 64;
#pragma unroll
      for (int t = 0; t < 8; ++t) {
        u16x8 v = *(const u16x8*)(vr + t * 8);
#pragma unroll
        for (int e = 0; e < 8; ++e) sVt[(lh * 64 + t * 8 + e) * 40 + lr] = v[e];
      }
    }
    __syncthreads();
    // ---- out += P (16x32) * V (32x128) over 8 dv tiles
    v16bf pf = frag_a(sP, 40);
#pragma unroll
    for (int j = 0; j < 8; ++j) {
      v16bf vf = frag_b(sVt + j * 16 * 40, 40);
      acc[j] = wmma_bf16(pf, vf, acc[j]);
    }
    __syncthreads();
  }
  // ---- sink logit joins the denominator; normalize; store bf16 [s][h*128+dv]
  const float sk = sinks[h];
  float inv[8];
#pragma unroll
  for (int r = 0; r < 8; ++r) inv[r] = 1.0f / (rsum[r] + __expf(sk - rmax[r]));
#pragma unroll
  for (int j = 0; j < 8; ++j)
#pragma unroll
    for (int r = 0; r < 8; ++r) {
      int m = q0 + r + lh * 8;
      int dv = j * 16 + lr;
      O[(size_t)m * (H_ * D_V) + h * D_V + dv] = f2bf(acc[j][r] * inv[r]);
    }
}

// ---------------------------------------------------------------------------
extern "C" void kernel_launch(void* const* d_in, const int* in_sizes, int n_in,
                              void* d_out, int out_size, void* d_ws, size_t ws_size,
                              hipStream_t stream) {
  const float* hidden = (const float*)d_in[0];
  const float* cosb   = (const float*)d_in[1];
  const float* sinb   = (const float*)d_in[2];
  const float* w_q    = (const float*)d_in[3];
  const float* b_q    = (const float*)d_in[4];
  const float* w_kva  = (const float*)d_in[5];
  const float* b_kva  = (const float*)d_in[6];
  const float* w_kvb  = (const float*)d_in[7];
  const float* w_o    = (const float*)d_in[8];
  const float* b_o    = (const float*)d_in[9];
  const float* sinks  = (const float*)d_in[10];

  char* ws = (char*)d_ws;
  size_t off = 0;
  auto alloc = [&](size_t bytes) {
    void* p = ws + off;
    off += (bytes + 255) & ~(size_t)255;
    return p;
  };
  float* bigf = (float*)alloc((size_t)S_ * 8192 * 4);   // q_f32, then kv_f32
  float* ckv  = (float*)alloc((size_t)S_ * (RANK_ + D_ROPE) * 4);
  float* krop = (float*)alloc((size_t)S_ * D_ROPE * 4);
  unsigned short* Qb    = (unsigned short*)alloc((size_t)H_ * S_ * D_QK * 2);
  unsigned short* Kb    = (unsigned short*)alloc((size_t)H_ * S_ * D_QK * 2);
  unsigned short* Vb    = (unsigned short*)alloc((size_t)H_ * S_ * D_V * 2);
  unsigned short* Ab    = (unsigned short*)alloc((size_t)S_ * H_ * D_V * 2);
  unsigned short* hb    = (unsigned short*)alloc((size_t)S_ * HID_ * 2);
  unsigned short* latb  = (unsigned short*)alloc((size_t)S_ * RANK_ * 2);
  unsigned short* wqT   = (unsigned short*)alloc((size_t)6144 * HID_ * 2);
  unsigned short* wkvaT = (unsigned short*)alloc((size_t)576 * HID_ * 2);
  unsigned short* wkvbT = (unsigned short*)alloc((size_t)8192 * RANK_ * 2);
  unsigned short* woT   = (unsigned short*)alloc((size_t)HID_ * HID_ * 2);
  (void)ws_size; (void)in_sizes; (void)n_in; (void)out_size;

  // 0) one-time bf16 conversion / weight transposition (amortized vs GEMM compute)
  cvt_bf16_k<<<(S_ * HID_ / 4) / 256, 256, 0, stream>>>(hidden, hb, HID_ / 4, HID_, HID_);
  transpose_cvt_k<<<dim3(6144 / 64, HID_ / 64), 256, 0, stream>>>(w_q, wqT, HID_, 6144);
  transpose_cvt_k<<<dim3(576 / 64, HID_ / 64), 256, 0, stream>>>(w_kva, wkvaT, HID_, 576);
  transpose_cvt_k<<<dim3(8192 / 64, RANK_ / 64), 256, 0, stream>>>(w_kvb, wkvbT, RANK_, 8192);
  transpose_cvt_k<<<dim3(HID_ / 64, HID_ / 64), 256, 0, stream>>>(w_o, woT, HID_, HID_);

  // 1) q = hidden @ w_q + b_q          [2048 x 6144]
  gemm_k<<<dim3(6144 / 128, S_ / 128), 256, 0, stream>>>(
      hb, wqT, b_q, bigf, S_, 6144, HID_, HID_, HID_, 6144);
  // 2) ckv = hidden @ w_kva + b_kva    [2048 x 576]
  gemm_k<<<dim3((576 + 127) / 128, S_ / 128), 256, 0, stream>>>(
      hb, wkvaT, b_kva, ckv, S_, 576, HID_, HID_, HID_, 576);
  // 3) RoPE + pack Q (consumes bigf)
  rope_pack_q_k<<<(S_ * H_ * D_QK) / 256, 256, 0, stream>>>(bigf, cosb, sinb, Qb);
  // 4) RoPE shared k_rope; convert latent to bf16
  rope_k_k<<<(S_ * D_ROPE) / 256, 256, 0, stream>>>(ckv, cosb, sinb, krop);
  cvt_bf16_k<<<(S_ * RANK_ / 4) / 256, 256, 0, stream>>>(ckv, latb, RANK_ / 4, RANK_ + D_ROPE, RANK_);
  // 5) kv = latent @ w_kvb             [2048 x 8192] (reuses bigf)
  gemm_k<<<dim3(8192 / 128, S_ / 128), 256, 0, stream>>>(
      latb, wkvbT, nullptr, bigf, S_, 8192, RANK_, RANK_, RANK_, 8192);
  // 6) pack K (nope + broadcast rope) and V, head-major bf16
  pack_kv_k<<<(H_ * S_ * 320) / 256, 256, 0, stream>>>(bigf, krop, Kb, Vb);
  // 7) sliding-window attention with sinks
  attn_k<<<dim3(S_ / 16, H_), 32, 0, stream>>>(Qb, Kb, Vb, sinks, Ab);
  // 8) out = attn @ w_o + b_o          [2048 x 4096] fp32
  gemm_k<<<dim3(4096 / 128, S_ / 128), 256, 0, stream>>>(
      Ab, woT, b_o, (float*)d_out, S_, 4096, H_ * D_V, H_ * D_V, H_ * D_V, 4096);
}
